// E8Attention_43224550867394
// MI455X (gfx1250) — matrix operand order repaired and checked by reading, not compiled, not run on hardware
//
#include <hip/hip_runtime.h>
#include <hip/hip_bf16.h>

typedef __attribute__((ext_vector_type(16))) __bf16 v16bf;
typedef __attribute__((ext_vector_type(8)))  __bf16 v8bf;
typedef __attribute__((ext_vector_type(8)))  float  v8f;

#define WMMA_BF16(a, b, c) \
    __builtin_amdgcn_wmma_f32_16x16x32_bf16(false, (a), false, (b), (short)0, (c), false, false)

constexpr int Bc  = 2;
constexpr int Tc  = 2048;
constexpr int Dc  = 1024;
constexpr int Hc  = 16;
constexpr int HDc = 64;
constexpr int BTc = 4096;
constexpr int Kc  = 1024;

__device__ inline v8f zero8() {
    v8f z;
    #pragma unroll
    for (int i = 0; i < 8; ++i) z[i] = 0.0f;
    return z;
}

__device__ inline v16bf joinv(v8bf lo, v8bf hi) {
    return __builtin_shufflevector(lo, hi, 0,1,2,3,4,5,6,7,8,9,10,11,12,13,14,15);
}

// A-matrix fragment (16x32 bf16, ISA 7.12.2): lane&15 = M row; lanes<16 hold
// K = {0..7, 16..23}, lanes>=16 hold K = {8..15, 24..31}.  Two 16B loads/lane.
__device__ inline v16bf loadA(const __bf16* p, int stride, int lane) {
    const __bf16* r = p + (size_t)(lane & 15) * stride + ((lane >> 4) << 3);
    return joinv(*(const v8bf*)r, *(const v8bf*)(r + 16));
}

// B-matrix fragment (32x16 bf16): lane&15 = N; lanes<16 hold K = {0..7, 16..23},
// lanes>=16 hold K = {8..15, 24..31} (same as A).  Source is [N][K] row-major.
__device__ inline v16bf loadB(const __bf16* p, int stride, int lane) {
    const __bf16* r = p + (size_t)(lane & 15) * stride + ((lane >> 4) << 3);
    return joinv(*(const v8bf*)r, *(const v8bf*)(r + 16));
}

// ---------------- conversion kernels ----------------
__global__ void cvt_bf16(const float* __restrict__ src, __bf16* __restrict__ dst, int n) {
    int i = blockIdx.x * blockDim.x + threadIdx.x;
    if (i < n) dst[i] = (__bf16)src[i];
}

// W [K=1024][N=1024] f32 -> Wt [N][K] bf16
__global__ void cvt_bf16_T(const float* __restrict__ src, __bf16* __restrict__ dst) {
    int i = blockIdx.x * blockDim.x + threadIdx.x;
    int k = i >> 10, n = i & 1023;
    dst[(size_t)n * 1024 + k] = (__bf16)src[i];
}

// ---------------- E8 table build ----------------
// roots[240][8]; rootsProj[240][64] = roots @ W_from_e8.  All 240 roots have
// squared norm exactly 2, so nearest-root reduces to argmax(dot).
__global__ void build_e8(const float* __restrict__ Wfrom,  // [8][64]
                         float* __restrict__ roots, float* __restrict__ rootsProj) {
    int tid = threadIdx.x;
    if (tid >= 240) return;
    float rt[8];
    if (tid < 112) {
        int p = tid >> 2, s = tid & 3;
        int i = 0, rem = p;
        while (rem >= 7 - i) { rem -= 7 - i; ++i; }
        int j = i + 1 + rem;
        #pragma unroll
        for (int d = 0; d < 8; ++d) rt[d] = 0.0f;
        rt[i] = (s & 2) ? -1.0f : 1.0f;
        rt[j] = (s & 1) ? -1.0f : 1.0f;
    } else {
        int n = tid - 112, mask = 0, cnt = -1;
        for (int m = 0; m < 256; ++m)
            if ((__popc(m) & 1) == 0) { ++cnt; if (cnt == n) { mask = m; break; } }
        #pragma unroll
        for (int d = 0; d < 8; ++d) rt[d] = ((mask >> d) & 1) ? -0.5f : 0.5f;
    }
    #pragma unroll
    for (int d = 0; d < 8; ++d) roots[tid * 8 + d] = rt[d];
    for (int c = 0; c < 64; ++c) {
        float acc = 0.0f;
        #pragma unroll
        for (int d = 0; d < 8; ++d) acc += rt[d] * Wfrom[d * 64 + c];
        rootsProj[tid * 64 + c] = acc;
    }
}

// ---------------- tiled bf16 GEMM, C = A[4096xK] * Bt[1024xK]^T ----------------
// MODE 0: f32 -> Q [b,h,t,d]   MODE 1: bf16 -> K [b,h,t,d]
// MODE 2: bf16 -> Vt [b,h,d,t] MODE 3: f32 -> plain [m][n]
// MODE is a template parameter so each instantiation has a straight-line epilogue.
template <int MODE>
__global__ __launch_bounds__(256) void gemm_bf16(
    const __bf16* __restrict__ A, const __bf16* __restrict__ Bt,
    void* __restrict__ Cout) {
    __shared__ alignas(16) __bf16 At[128 * 40];   // 80B rows: conflict-free
    __shared__ alignas(16) __bf16 Bs[128 * 40];

    const int tid  = threadIdx.x;
    const int lane = tid & 31;
    const int w    = tid >> 5;
    const int wrow = (w & 3) * 32;   // wave M offset in tile
    const int wcol = (w >> 2) * 64;  // wave N offset in tile
    const int tm   = blockIdx.y * 128;
    const int tn   = blockIdx.x * 128;

    v8f acc[2][4];
    #pragma unroll
    for (int i = 0; i < 2; ++i)
        #pragma unroll
        for (int j = 0; j < 4; ++j) acc[i][j] = zero8();

    const int lrow  = tid >> 1;
    const int lhalf = (tid & 1) << 4;  // 0 or 16 elements

    const __bf16* ga = A  + (size_t)(tm + lrow) * Kc + lhalf;
    const __bf16* gb = Bt + (size_t)(tn + lrow) * Kc + lhalf;

    for (int kb = 0; kb < Kc; kb += 32) {
        {   // stage 128x32 A and B tiles (coalesced 16B loads)
            v8bf a0 = *(const v8bf*)(ga + kb);
            v8bf a1 = *(const v8bf*)(ga + kb + 8);
            v8bf b0 = *(const v8bf*)(gb + kb);
            v8bf b1 = *(const v8bf*)(gb + kb + 8);
            // prefetch next K-step tile rows into cache (global_prefetch_b8)
            if (kb + 32 < Kc) {
                __builtin_prefetch(ga + kb + 32, 0, 3);
                __builtin_prefetch(gb + kb + 32, 0, 3);
            }
            *(v8bf*)(At + lrow * 40 + lhalf)     = a0;
            *(v8bf*)(At + lrow * 40 + lhalf + 8) = a1;
            *(v8bf*)(Bs + lrow * 40 + lhalf)     = b0;
            *(v8bf*)(Bs + lrow * 40 + lhalf + 8) = b1;
        }
        __syncthreads();
        v16bf af[2], bf[4];
        #pragma unroll
        for (int i = 0; i < 2; ++i) af[i] = loadA(At + (wrow + i * 16) * 40, 40, lane);
        #pragma unroll
        for (int j = 0; j < 4; ++j) bf[j] = loadB(Bs + (wcol + j * 16) * 40, 40, lane);
        #pragma unroll
        for (int i = 0; i < 2; ++i)
            #pragma unroll
            for (int j = 0; j < 4; ++j)
                acc[i][j] = WMMA_BF16(af[i], bf[j], acc[i][j]);
        __syncthreads();
    }

    // epilogue: acc VGPR r <-> row r (lanes 0-15) / r+8 (lanes 16-31), N = lane&15
    const int mhalf = (lane >> 4) * 8;
    const int nloc  = lane & 15;
    const int bidx  = tm >> 11;          // block-uniform: tile never straddles a batch
    const int tbase = (tm & 2047) + wrow + mhalf;
    #pragma unroll
    for (int i = 0; i < 2; ++i)
        #pragma unroll
        for (int j = 0; j < 4; ++j) {
            const int n = tn + wcol + j * 16 + nloc;
            const int h = n >> 6, d = n & 63;
            #pragma unroll
            for (int r = 0; r < 8; ++r) {
                const int t = tbase + i * 16 + r;
                const float v = acc[i][j][r];
                if (MODE == 3) {
                    ((float*)Cout)[((size_t)(tm + wrow + mhalf + i * 16 + r)) * Dc + n] = v;
                } else {
                    const size_t bh = (size_t)bidx * Hc + h;
                    if (MODE == 0)
                        ((float*)Cout)[(bh * Tc + t) * HDc + d] = v;
                    else if (MODE == 1)
                        ((__bf16*)Cout)[(bh * Tc + t) * HDc + d] = (__bf16)v;
                    else
                        ((__bf16*)Cout)[(bh * HDc + d) * Tc + t] = (__bf16)v;
                }
            }
        }
}

// ---------------- E8 quantization of Q ----------------
__global__ __launch_bounds__(256) void e8_quant(
    const float* __restrict__ Qf, const float* __restrict__ Wte,     // [64][8]
    const float* __restrict__ roots, const float* __restrict__ rootsProj,
    __bf16* __restrict__ Qout) {
    __shared__ float sW[64 * 8];
    __shared__ float sR[240 * 8];
    const int tid = threadIdx.x;
    for (int i = tid; i < 512; i += 256)  sW[i] = Wte[i];
    for (int i = tid; i < 1920; i += 256) sR[i] = roots[i];
    __syncthreads();

    const size_t row = (size_t)blockIdx.x * 256 + tid;  // 65536 rows
    const float* q = Qf + row * 64;
    float y[8] = {0, 0, 0, 0, 0, 0, 0, 0};
    for (int d = 0; d < 64; ++d) {
        float qd = q[d];
        #pragma unroll
        for (int j = 0; j < 8; ++j) y[j] += qd * sW[d * 8 + j];
    }
    float best = -1e30f; int bi = 0;
    for (int i = 0; i < 240; ++i) {
        const float* r = sR + i * 8;
        float s = y[0]*r[0] + y[1]*r[1] + y[2]*r[2] + y[3]*r[3]
                + y[4]*r[4] + y[5]*r[5] + y[6]*r[6] + y[7]*r[7];
        if (s > best) { best = s; bi = i; }
    }
    const float* pr = rootsProj + bi * 64;
    __bf16* o = Qout + row * 64;
    for (int d = 0; d < 64; ++d) o[d] = (__bf16)pr[d];
}

// ---------------- causal flash attention ----------------
// grid.x = T/64 query tiles, grid.y = B*H.  4 waves x 16 queries each.
__global__ __launch_bounds__(128) void attn_kernel(
    const __bf16* __restrict__ Qbf, const __bf16* __restrict__ Kbf,
    const __bf16* __restrict__ Vt, __bf16* __restrict__ Obf) {
    __shared__ alignas(16) __bf16 pTile[4][16 * 40];  // per-wave P staging

    const int tid  = threadIdx.x;
    const int lane = tid & 31;
    const int w    = tid >> 5;
    const int bh   = blockIdx.y;
    const int b    = bh >> 4, h = bh & 15;
    const int qt   = blockIdx.x * 64;
    const int q0   = qt + w * 16;

    const __bf16* Qp = Qbf + ((size_t)bh * Tc + q0) * HDc;
    const __bf16* Kp = Kbf + (size_t)bh * Tc * HDc;
    const __bf16* Vp = Vt  + (size_t)bh * HDc * Tc;

    v16bf qf0 = loadA(Qp, HDc, lane);        // hd 0..31
    v16bf qf1 = loadA(Qp + 32, HDc, lane);   // hd 32..63

    float mrow[8], lrow[8];
    v8f o[4];
    #pragma unroll
    for (int r = 0; r < 8; ++r) { mrow[r] = -1e30f; lrow[r] = 0.0f; }
    #pragma unroll
    for (int n = 0; n < 4; ++n) o[n] = zero8();

    const int   mhalf = (lane >> 4) * 8;
    const int   nloc  = lane & 15;
    const int   kend  = qt + 64;             // uniform trip count across block
    const float scale = 0.125f;              // 1/sqrt(64)

    for (int kt = 0; kt < kend; kt += 32) {
        // S tiles: keys kt..kt+15 (s0) and kt+16..kt+31 (s1), hd contraction = 64
        v8f s0 = zero8(), s1 = zero8();
        {
            v16bf ka0 = loadB(Kp + (size_t)kt * HDc,             HDc, lane);
            v16bf ka1 = loadB(Kp + (size_t)kt * HDc + 32,        HDc, lane);
            v16bf kb0 = loadB(Kp + (size_t)(kt + 16) * HDc,      HDc, lane);
            v16bf kb1 = loadB(Kp + (size_t)(kt + 16) * HDc + 32, HDc, lane);
            s0 = WMMA_BF16(qf0, ka0, s0);
            s0 = WMMA_BF16(qf1, ka1, s0);
            s1 = WMMA_BF16(qf0, kb0, s1);
            s1 = WMMA_BF16(qf1, kb1, s1);
        }
        // online softmax: row stats via 16-lane butterfly (rows r / r+8 per half)
        #pragma unroll
        for (int r = 0; r < 8; ++r) {
            int q  = q0 + r + mhalf;
            float a  = ((kt + nloc)      <= q) ? s0[r] * scale : -1.0e9f;
            float bb = ((kt + 16 + nloc) <= q) ? s1[r] * scale : -1.0e9f;
            float v = fmaxf(a, bb);
            #pragma unroll
            for (int off = 1; off < 16; off <<= 1) v = fmaxf(v, __shfl_xor(v, off, 32));
            float mnew = fmaxf(mrow[r], v);
            float fr   = __expf(mrow[r] - mnew);
            mrow[r] = mnew;
            float p0 = __expf(a - mnew);
            float p1 = __expf(bb - mnew);
            float rs = p0 + p1;
            #pragma unroll
            for (int off = 1; off < 16; off <<= 1) rs += __shfl_xor(rs, off, 32);
            lrow[r] = lrow[r] * fr + rs;
            #pragma unroll
            for (int n = 0; n < 4; ++n) o[n][r] *= fr;
            pTile[w][(r + mhalf) * 40 + nloc]      = (__bf16)p0;
            pTile[w][(r + mhalf) * 40 + nloc + 16] = (__bf16)p1;
        }
        __syncthreads();                       // publish P (cross-lane relayout)
        v16bf pf = loadA(pTile[w], 40, lane);  // P as A-fragment (16q x 32k)
        #pragma unroll
        for (int n = 0; n < 4; ++n) {
            v16bf vf = loadB(Vp + (size_t)(n * 16) * Tc + kt, Tc, lane);
            o[n] = WMMA_BF16(pf, vf, o[n]);
        }
        __syncthreads();                       // protect pTile for next iter
    }

    // normalize and store O as bf16 in [b][t][h*64+d] layout for the final GEMM
    #pragma unroll
    for (int n = 0; n < 4; ++n)
        #pragma unroll
        for (int r = 0; r < 8; ++r) {
            int q = q0 + r + mhalf;
            int c = h * HDc + n * 16 + nloc;
            float v = o[n][r] / lrow[r];
            Obf[((size_t)b * Tc + q) * Dc + c] = (__bf16)v;
        }
}

// ---------------- host side ----------------
extern "C" void kernel_launch(void* const* d_in, const int* in_sizes, int n_in,
                              void* d_out, int out_size, void* d_ws, size_t ws_size,
                              hipStream_t stream) {
    const float* x     = (const float*)d_in[0];
    const float* Wq    = (const float*)d_in[1];
    const float* Wk    = (const float*)d_in[2];
    const float* Wv    = (const float*)d_in[3];
    const float* Wte   = (const float*)d_in[4];
    const float* Wfrom = (const float*)d_in[5];
    const float* Wout  = (const float*)d_in[6];
    float* out = (float*)d_out;

    char* ws = (char*)d_ws;
    const size_t MB = 1024 * 1024;
    __bf16* Xbf   = (__bf16*)(ws);             //  8 MB  x bf16
    __bf16* Wqt   = (__bf16*)(ws +  8 * MB);   //  2 MB  Wq^T bf16
    __bf16* Wkt   = (__bf16*)(ws + 10 * MB);
    __bf16* Wvt   = (__bf16*)(ws + 12 * MB);
    __bf16* Wot   = (__bf16*)(ws + 14 * MB);
    float*  Qf    = (float*) (ws + 16 * MB);   // 16 MB  Q f32 [b,h,t,d]
    __bf16* Kbf   = (__bf16*)(ws + 32 * MB);   //  8 MB  K bf16 [b,h,t,d]
    __bf16* Vtb   = (__bf16*)(ws + 40 * MB);   //  8 MB  V bf16 [b,h,d,t]
    __bf16* Qbf   = (__bf16*)(ws + 48 * MB);   //  8 MB  quantized Q bf16
    __bf16* Obf   = (__bf16*)(ws + 56 * MB);   //  8 MB  attention out bf16
    float*  roots = (float*) (ws + 64 * MB);
    float*  rProj = (float*) (ws + 64 * MB + 16 * 1024);

    const int nX = BTc * Dc;   // 4194304
    const int nW = Dc * Dc;    // 1048576

    cvt_bf16  <<<(nX + 255) / 256, 256, 0, stream>>>(x, Xbf, nX);
    cvt_bf16_T<<<nW / 256, 256, 0, stream>>>(Wq,   Wqt);
    cvt_bf16_T<<<nW / 256, 256, 0, stream>>>(Wk,   Wkt);
    cvt_bf16_T<<<nW / 256, 256, 0, stream>>>(Wv,   Wvt);
    cvt_bf16_T<<<nW / 256, 256, 0, stream>>>(Wout, Wot);
    build_e8  <<<1, 256, 0, stream>>>(Wfrom, roots, rProj);

    dim3 gg(Dc / 128, BTc / 128);  // (8, 32)
    gemm_bf16<0><<<gg, 256, 0, stream>>>(Xbf, Wqt, Qf);
    gemm_bf16<1><<<gg, 256, 0, stream>>>(Xbf, Wkt, Kbf);
    gemm_bf16<2><<<gg, 256, 0, stream>>>(Xbf, Wvt, Vtb);

    e8_quant<<<256, 256, 0, stream>>>(Qf, Wte, roots, rProj, Qbf);

    attn_kernel<<<dim3(Tc / 64, Bc * Hc), 128, 0, stream>>>(Qbf, Kbf, Vtb, Obf);

    gemm_bf16<3><<<gg, 256, 0, stream>>>(Obf, Wot, out);
}
